// HSTU_82360292868853
// MI455X (gfx1250) — matrix operand order repaired and checked by reading, not compiled, not run on hardware
//
#include <hip/hip_runtime.h>
#include <hip/hip_bf16.h>
#include <stdint.h>

// ---------------- problem constants ----------------
#define BDIM   4
#define SEQ    2048
#define DMODEL 1024
#define NHEAD  8
#define ADIM   128
#define LVDIM  128
#define QKVW   4096          // 2*(A+Lv)*H columns in the stored weight
#define QKVP   3072          // packed: only q,k,v (384 per head) are used
#define MROWS  (BDIM*SEQ)    // 8192

typedef __attribute__((ext_vector_type(16))) __bf16 v16bf;
typedef __attribute__((ext_vector_type(8)))  float  v8f;

union FragBF { v16bf v; uint4 q[2]; };

__device__ __forceinline__ unsigned short f2bf(float f) {
  union { float f; unsigned u; } c; c.f = f;
  unsigned u = c.u;
  unsigned r = (u + 0x7fffu + ((u >> 16) & 1u)) >> 16;   // round-nearest-even
  return (unsigned short)r;
}

__device__ __forceinline__ v8f vzero8() {
  v8f z;
#pragma unroll
  for (int t = 0; t < 8; ++t) z[t] = 0.0f;
  return z;
}

// ---------------- weight convert + pack + transpose (f32 -> bf16) ----------------
// qwb[c][d] = qkv_w[d][h*512 + w]  (c = h*384 + w, w<384)   -> B^T layout (N x K)
// owb[n][k] = out_w[k][n]                                   -> B^T layout (N x K)
__global__ __launch_bounds__(256) void convw_kernel(
    const float* __restrict__ qw, const float* __restrict__ ow,
    unsigned short* __restrict__ qwb, unsigned short* __restrict__ owb)
{
  int idx = blockIdx.x * 256 + threadIdx.x;
  const int T1 = QKVP * DMODEL;
  if (idx < T1) {
    int d = idx / QKVP;
    int c = idx - d * QKVP;
    int h = c / 384;
    int w = c - h * 384;
    qwb[(size_t)c * DMODEL + d] = f2bf(qw[(size_t)d * QKVW + h * 512 + w]);
  } else {
    int j = idx - T1;
    int k = j >> 10;
    int n = j & (DMODEL - 1);
    owb[(size_t)n * DMODEL + k] = f2bf(ow[(size_t)k * DMODEL + n]);
  }
}

// ---------------- layernorm + cast to bf16 ----------------
__global__ __launch_bounds__(256) void ln_kernel(
    const float* __restrict__ x, const float* __restrict__ g,
    const float* __restrict__ bb, unsigned short* __restrict__ xn)
{
  __shared__ float r1[256], r2[256];
  const int row = blockIdx.x;
  const int tid = threadIdx.x;
  const float4 v = ((const float4*)(x + (size_t)row * DMODEL))[tid];
  r1[tid] = v.x + v.y + v.z + v.w;
  r2[tid] = v.x*v.x + v.y*v.y + v.z*v.z + v.w*v.w;
  __syncthreads();
  for (int st = 128; st > 0; st >>= 1) {
    if (tid < st) { r1[tid] += r1[tid + st]; r2[tid] += r2[tid + st]; }
    __syncthreads();
  }
  const float mu  = r1[0] * (1.0f / DMODEL);
  const float var = r2[0] * (1.0f / DMODEL) - mu * mu;
  const float rs  = rsqrtf(var + 1e-5f);
  const int c = tid * 4;
  const float4 gv = *(const float4*)(g + c);
  const float4 bv = *(const float4*)(bb + c);
  alignas(8) unsigned short o[4];
  o[0] = f2bf((v.x - mu) * rs * gv.x + bv.x);
  o[1] = f2bf((v.y - mu) * rs * gv.y + bv.y);
  o[2] = f2bf((v.z - mu) * rs * gv.z + bv.z);
  o[3] = f2bf((v.w - mu) * rs * gv.w + bv.w);
  *(uint2*)&xn[(size_t)row * DMODEL + c] = *(const uint2*)o;
}

// ---------------- tiled bf16 WMMA GEMM, 128x128 tile, K=1024 ----------------
// A:  M x 1024 bf16 row-major.   Bt: N x 1024 bf16 (i.e. B transposed).
// MODE 0: qkv epilogue  (add bias, scatter to q/k/v^T bf16 buffers)
// MODE 1: out epilogue  (out[m][n] += acc + bias[n], f32 residual in-place)
template<int MODE>
__global__ __launch_bounds__(256) void wmma_gemm_kernel(
    const unsigned short* __restrict__ Ag,
    const unsigned short* __restrict__ Btg,
    const float* __restrict__ bias,
    unsigned short* __restrict__ qb,
    unsigned short* __restrict__ kbuf,
    unsigned short* __restrict__ vbuf,
    float* outf)
{
  constexpr int KD = DMODEL, LDK = 40;            // padded LDS stride (banks)
  __shared__ unsigned short As[128 * LDK];
  __shared__ unsigned short Bs[128 * LDK];

  const int tid  = threadIdx.x;
  const int lane = tid & 31;
  const int wid  = tid >> 5;
  const int wm   = wid & 3;          // 4 M-strips of 32 rows
  const int wn   = wid >> 2;         // 2 N-strips of 64 cols
  const int m0   = blockIdx.y * 128;
  const int n0   = blockIdx.x * 128;
  const int lh   = lane >> 4, ll = lane & 15;

  v8f acc[2][4];
#pragma unroll
  for (int a = 0; a < 2; ++a)
#pragma unroll
    for (int bI = 0; bI < 4; ++bI) acc[a][bI] = vzero8();

  for (int k0 = 0; k0 < KD; k0 += 32) {
#pragma unroll
    for (int rep = 0; rep < 2; ++rep) {           // 512 16B-chunks / 256 thr
      int ci  = tid + rep * 256;
      int row = ci >> 2;
      int kk  = (ci & 3) * 8;
      *(uint4*)&As[row * LDK + kk] = *(const uint4*)&Ag [(size_t)(m0 + row) * KD + k0 + kk];
      *(uint4*)&Bs[row * LDK + kk] = *(const uint4*)&Btg[(size_t)(n0 + row) * KD + k0 + kk];
    }
    __syncthreads();

    FragBF af[2], bf[4];
#pragma unroll
    for (int mi = 0; mi < 2; ++mi) {              // A frag: row ll, K chunks 8*lh, +16
      int r = wm * 32 + mi * 16 + ll;
      af[mi].q[0] = *(const uint4*)&As[r * LDK + 8 * lh];
      af[mi].q[1] = *(const uint4*)&As[r * LDK + 8 * lh + 16];
    }
#pragma unroll
    for (int ni = 0; ni < 4; ++ni) {              // B frag: col ll, K chunks 8*lh, +16
      int c = wn * 64 + ni * 16 + ll;
      bf[ni].q[0] = *(const uint4*)&Bs[c * LDK + 8 * lh];
      bf[ni].q[1] = *(const uint4*)&Bs[c * LDK + 8 * lh + 16];
    }
#pragma unroll
    for (int mi = 0; mi < 2; ++mi)
#pragma unroll
      for (int ni = 0; ni < 4; ++ni)
        acc[mi][ni] = __builtin_amdgcn_wmma_f32_16x16x32_bf16(
            false, af[mi].v, false, bf[ni].v, (short)0, acc[mi][ni], false, false);
    __syncthreads();
  }

#pragma unroll
  for (int mi = 0; mi < 2; ++mi)
#pragma unroll
    for (int ni = 0; ni < 4; ++ni)
#pragma unroll
      for (int i = 0; i < 8; ++i) {
        const int m = m0 + wm * 32 + mi * 16 + i + 8 * lh;   // D layout: M=i(+8), N=ll
        const int n = n0 + wn * 64 + ni * 16 + ll;
        float v = acc[mi][ni][i];
        if (MODE == 0) {
          const int b  = m >> 11;
          const int nn = m & (SEQ - 1);
          const int h  = n / 384;
          const int w  = n - h * 384;
          v += bias[h * 512 + w];
          const int bh = b * NHEAD + h;
          const unsigned short bv = f2bf(v);
          if (w < 128)        qb  [((size_t)(bh * SEQ + nn)) * ADIM + w]           = bv;
          else if (w < 256)   kbuf[((size_t)(bh * SEQ + nn)) * ADIM + (w - 128)]   = bv;
          else                vbuf[((size_t)(bh * LVDIM + (w - 256))) * SEQ + nn]  = bv; // V^T
        } else {
          const size_t idx = (size_t)m * DMODEL + n;
          outf[idx] = outf[idx] + v + bias[n];               // residual in-place
        }
      }
}

// ---------------- fused attention: scores -> silu/N -> exp -> P@V ----------------
// 4 waves x 16 query rows per workgroup; one (b,h) per blockIdx.y.
__global__ __launch_bounds__(128) void attn_kernel(
    const unsigned short* __restrict__ qb,
    const unsigned short* __restrict__ kbuf,
    const unsigned short* __restrict__ vbuf,
    const int* __restrict__ past_ids,
    unsigned short* __restrict__ attn)
{
  constexpr int PLD = 40;
  __shared__ unsigned short Pb[4][16 * PLD];     // per-wave P tile (16x32 bf16)
  __shared__ float Dsh[4][16][17];
  __shared__ float Drow[4][16];
  __shared__ unsigned char vK[SEQ];

  const int tid  = threadIdx.x;
  const int lane = tid & 31;
  const int wid  = tid >> 5;
  const int lh   = lane >> 4, ll = lane & 15;
  const int bh   = blockIdx.y;
  const int b    = bh >> 3;
  const int h    = bh & 7;
  const int q0   = blockIdx.x * 64 + wid * 16;

  for (int k = tid; k < SEQ; k += 128)
    vK[k] = (past_ids[b * SEQ + k] != 0) ? 1 : 0;
  __syncthreads();

  unsigned vq8 = 0;                               // validity of my 8 row slots
#pragma unroll
  for (int i = 0; i < 8; ++i)
    vq8 |= (unsigned)vK[q0 + i + 8 * lh] << i;

  FragBF qf[4];                                   // 16x128 q rows, 4 K-chunks
  {
    const size_t base = ((size_t)(bh * SEQ + q0 + ll)) * ADIM;
#pragma unroll
    for (int c = 0; c < 4; ++c) {
      qf[c].q[0] = *(const uint4*)&qb[base + c * 32 + 8 * lh];
      qf[c].q[1] = *(const uint4*)&qb[base + c * 32 + 8 * lh + 16];
    }
  }

  v8f oacc[8];
#pragma unroll
  for (int j = 0; j < 8; ++j) oacc[j] = vzero8();
  v8f den = vzero8();

  const float scl  = 0.08838834764831845f;        // 1/sqrt(128)
  const float invN = 1.0f / (float)SEQ;

  for (int k0 = 0; k0 < SEQ; k0 += 32) {
#pragma unroll
    for (int sub = 0; sub < 2; ++sub) {
      const int kk0 = k0 + sub * 16;
      v8f sacc = vzero8();
      const size_t kbase = ((size_t)(bh * SEQ + kk0 + ll)) * ADIM + 8 * lh;
#pragma unroll
      for (int c = 0; c < 4; ++c) {               // B frag = K rows (already B^T)
        FragBF kf;
        kf.q[0] = *(const uint4*)&kbuf[kbase + c * 32];
        kf.q[1] = *(const uint4*)&kbuf[kbase + c * 32 + 16];
        sacc = __builtin_amdgcn_wmma_f32_16x16x32_bf16(
            false, qf[c].v, false, kf.v, (short)0, sacc, false, false);
      }
      const int kvalid = vK[kk0 + ll];
#pragma unroll
      for (int i = 0; i < 8; ++i) {
        const int m  = i + 8 * lh;
        const int nq = q0 + m;
        const int mk = kk0 + ll;
        const bool un = (mk > nq) && kvalid && ((vq8 >> i) & 1);
        const float s   = sacc[i] * scl;
        const float sig = 1.0f / (1.0f + __expf(-s));
        const float lg  = s * sig * invN;         // silu(s)/N  (tiny; no max needed)
        const float ev  = un ? __expf(lg) : 1e-20f; // masked->TINY: exact uniform fallback
        den[i] += ev;
        Pb[wid][m * PLD + sub * 16 + ll] = f2bf(ev);
      }
    }
    // P (16x32) @ V (32x128): A frag from LDS transpose, B frag from V^T global
    FragBF pf;
    pf.q[0] = *(const uint4*)&Pb[wid][ll * PLD + 8 * lh];
    pf.q[1] = *(const uint4*)&Pb[wid][ll * PLD + 8 * lh + 16];
#pragma unroll
    for (int j = 0; j < 8; ++j) {
      FragBF vf;
      const size_t vb = ((size_t)(bh * LVDIM + j * 16 + ll)) * SEQ + k0 + 8 * lh;
      vf.q[0] = *(const uint4*)&vbuf[vb];
      vf.q[1] = *(const uint4*)&vbuf[vb + 16];
      oacc[j] = __builtin_amdgcn_wmma_f32_16x16x32_bf16(
          false, pf.v, false, vf.v, (short)0, oacc[j], false, false);
    }
  }

  // denominator row-sum (16 lanes of one tile half hold one row's partials)
#pragma unroll
  for (int i = 0; i < 8; ++i) Dsh[wid][i + 8 * lh][ll] = den[i];
  if (lane < 16) {
    float s = 0.f;
#pragma unroll
    for (int n = 0; n < 16; ++n) s += Dsh[wid][lane][n];
    Drow[wid][lane] = s;
  }
#pragma unroll
  for (int j = 0; j < 8; ++j)
#pragma unroll
    for (int i = 0; i < 8; ++i) {
      const int m = i + 8 * lh;
      const float o = oacc[j][i] / Drow[wid][m];
      const size_t idx = ((size_t)(b * SEQ + q0 + m)) * (NHEAD * LVDIM)
                         + h * LVDIM + j * 16 + ll;
      attn[idx] = f2bf(o);
    }
}

// ---------------- host side ----------------
extern "C" void kernel_launch(void* const* d_in, const int* in_sizes, int n_in,
                              void* d_out, int out_size, void* d_ws, size_t ws_size,
                              hipStream_t stream)
{
  (void)in_sizes; (void)n_in; (void)out_size; (void)ws_size;
  const int*   past_ids = (const int*)  d_in[1];
  const float* past_emb = (const float*)d_in[2];
  const float* qkv_w    = (const float*)d_in[3];
  const float* qkv_b    = (const float*)d_in[4];
  const float* out_w    = (const float*)d_in[5];
  const float* out_b    = (const float*)d_in[6];
  const float* ln_g     = (const float*)d_in[7];
  const float* ln_bv    = (const float*)d_in[8];
  float* xout = (float*)d_out;

  char* w = (char*)d_ws;
  size_t off = 0;
  auto alloc = [&](size_t bytes) {
    void* p = w + off;
    off = (off + bytes + 255) & ~(size_t)255;
    return p;
  };
  unsigned short* xn  = (unsigned short*)alloc((size_t)MROWS * DMODEL * 2);
  unsigned short* qwb = (unsigned short*)alloc((size_t)QKVP  * DMODEL * 2);
  unsigned short* owb = (unsigned short*)alloc((size_t)DMODEL* DMODEL * 2);
  unsigned short* qb  = (unsigned short*)alloc((size_t)BDIM * NHEAD * SEQ * ADIM  * 2);
  unsigned short* kb  = (unsigned short*)alloc((size_t)BDIM * NHEAD * SEQ * ADIM  * 2);
  unsigned short* vb  = (unsigned short*)alloc((size_t)BDIM * NHEAD * SEQ * LVDIM * 2);
  unsigned short* at  = (unsigned short*)alloc((size_t)MROWS * DMODEL * 2);

  // residual stream x lives in d_out
  hipMemcpyAsync(xout, past_emb, (size_t)MROWS * DMODEL * sizeof(float),
                 hipMemcpyDeviceToDevice, stream);

  for (int i = 0; i < 4; ++i) {
    convw_kernel<<<16384, 256, 0, stream>>>(
        qkv_w + (size_t)i * DMODEL * QKVW,
        out_w + (size_t)i * DMODEL * DMODEL, qwb, owb);

    ln_kernel<<<MROWS, 256, 0, stream>>>(
        xout, ln_g + (size_t)i * DMODEL, ln_bv + (size_t)i * DMODEL, xn);

    wmma_gemm_kernel<0><<<dim3(QKVP / 128, MROWS / 128), 256, 0, stream>>>(
        xn, qwb, qkv_b + (size_t)i * QKVW, qb, kb, vb, nullptr);

    attn_kernel<<<dim3(SEQ / 64, BDIM * NHEAD), 128, 0, stream>>>(
        qb, kb, vb, past_ids, at);

    wmma_gemm_kernel<1><<<dim3(DMODEL / 128, MROWS / 128), 256, 0, stream>>>(
        at, owb, out_b + (size_t)i * DMODEL, nullptr, nullptr, nullptr, xout);
  }
}
